// MultiHeadAttention_71820443124240
// MI455X (gfx1250) — compile-verified, matrix operand order checked by reading, not run on hardware
//
#include <hip/hip_runtime.h>

// ---------------------------------------------------------------------------
// MI455X (gfx1250) multi-head attention forward.
//
// Roofline: ~69 GFLOP vs ~150MB traffic => compute bound. All matmuls are
// done as v_wmma_f32_16x16x32_bf16 (bf16 in, fp32 accumulate), which is the
// deep-K matrix path CDNA5 rewards; fp32 WMMA would only give K=4 per op.
// ---------------------------------------------------------------------------

typedef __attribute__((ext_vector_type(16))) __bf16          v16bf;
typedef __attribute__((ext_vector_type(8)))  float           v8f;
typedef __attribute__((ext_vector_type(4)))  unsigned int    u32x4;

#define D_MODEL   1024
#define NUM_HEADS 16
#define HEAD_DIM  64
#define SEQ       2048
#define BATCH     2
#define M_TOTAL   (BATCH * SEQ)        // 4096
#define N_QKV     (3 * D_MODEL)        // 3072

__device__ __forceinline__ unsigned short f2bf(float f) {
  unsigned int u = __float_as_uint(f);
  unsigned int r = u + 0x7FFFu + ((u >> 16) & 1u);   // round-to-nearest-even
  return (unsigned short)(r >> 16);
}

// Load a 16-element bf16 fragment for the ISA 16-bit A/B layout:
// per lane, elements 0..7 are 8 contiguous K values, elements 8..15 are the
// next 8 contiguous K values 16 further on. Caller passes p = row + 8*(lane>>4)
// (plus any K-chunk base), so this is two aligned 16-byte loads.
__device__ __forceinline__ v16bf frag8x2(const unsigned short* p) {
  union { v16bf v; u32x4 q[2]; } u;
  u.q[0] = *reinterpret_cast<const u32x4*>(p);
  u.q[1] = *reinterpret_cast<const u32x4*>(p + 16);
  return u.v;
}

__device__ __forceinline__ v8f wmma_bf16(v16bf a, v16bf b, v8f c) {
  return __builtin_amdgcn_wmma_f32_16x16x32_bf16(
      false, a, false, b, (short)0, c, false, false);
}

// ---------------------------------------------------------------------------
// fp32 -> bf16 elementwise convert
// ---------------------------------------------------------------------------
__global__ void f32_to_bf16_kernel(const float* __restrict__ in,
                                   unsigned short* __restrict__ out, int n) {
  int i = blockIdx.x * blockDim.x + threadIdx.x;
  if (i < n) out[i] = f2bf(in[i]);
}

// ---------------------------------------------------------------------------
// GEMM: C[M,N] = A[M,K] (bf16) * B[K,N] (bf16) + bias[N], fp32 out.
// Block = 256 threads (8 waves). Block tile 128x64, wave tile 32x32 (2x2 WMMA),
// K staged in LDS in chunks of 32. B staged transposed so both operand
// fragments are two ds_load_b128 per lane.
// ---------------------------------------------------------------------------
__global__ __launch_bounds__(256) void gemm_bf16_wmma(
    const unsigned short* __restrict__ A, const unsigned short* __restrict__ B,
    const float* __restrict__ bias, float* __restrict__ C,
    int M, int N, int K) {
  __shared__ unsigned short As[128][40];  // [m][k], padded, 16B-aligned rows
  __shared__ unsigned short Bs[64][40];   // transposed: [n][k]

  const int tid  = threadIdx.x;
  const int lane = tid & 31;
  const int wave = tid >> 5;
  const int wm   = wave & 3;   // 0..3 -> 32-row slabs
  const int wn   = wave >> 2;  // 0..1 -> 32-col slabs
  const int lg   = lane >> 4;  // half-wave group
  const int lr   = lane & 15;

  const int block_m = blockIdx.y * 128;
  const int block_n = blockIdx.x * 64;

  v8f acc[2][2];
#pragma unroll
  for (int i = 0; i < 2; ++i)
#pragma unroll
    for (int j = 0; j < 2; ++j) acc[i][j] = (v8f)0.0f;

  const int arow = tid >> 1;             // A stage: 128 rows x 32 cols
  const int acol = (tid & 1) * 16;
  const int bk   = tid >> 3;             // B stage: 32 rows x 64 cols
  const int bn   = (tid & 7) * 8;

  for (int k0 = 0; k0 < K; k0 += 32) {
    // Stage A tile (row-major)
    {
      const unsigned short* g = A + (size_t)(block_m + arow) * K + k0 + acol;
      u32x4 v0 = *reinterpret_cast<const u32x4*>(g);
      u32x4 v1 = *reinterpret_cast<const u32x4*>(g + 8);
      *reinterpret_cast<u32x4*>(&As[arow][acol])     = v0;
      *reinterpret_cast<u32x4*>(&As[arow][acol + 8]) = v1;
      if (k0 + 32 < K) __builtin_prefetch((const void*)(g + 32), 0, 1);
    }
    // Stage B tile transposed: Bs[n][k]
    {
      const unsigned short* g = B + (size_t)(k0 + bk) * N + block_n + bn;
      union { u32x4 q; unsigned short us[8]; } v;
      v.q = *reinterpret_cast<const u32x4*>(g);
#pragma unroll
      for (int j = 0; j < 8; ++j) Bs[bn + j][bk] = v.us[j];
      if (k0 + 32 < K) __builtin_prefetch((const void*)(g + 32 * N), 0, 1);
    }
    __syncthreads();

    v16bf af[2], bf[2];
#pragma unroll
    for (int mt = 0; mt < 2; ++mt)
      af[mt] = frag8x2(&As[wm * 32 + mt * 16 + lr][8 * lg]);
#pragma unroll
    for (int nt = 0; nt < 2; ++nt)
      bf[nt] = frag8x2(&Bs[wn * 32 + nt * 16 + lr][8 * lg]);
#pragma unroll
    for (int mt = 0; mt < 2; ++mt)
#pragma unroll
      for (int nt = 0; nt < 2; ++nt)
        acc[mt][nt] = wmma_bf16(af[mt], bf[nt], acc[mt][nt]);
    __syncthreads();
  }

  // Epilogue: C/D layout is VGPR r -> M = r + 8*(lane>>4), N = lane&15.
#pragma unroll
  for (int mt = 0; mt < 2; ++mt) {
#pragma unroll
    for (int nt = 0; nt < 2; ++nt) {
      const int gm = block_m + wm * 32 + mt * 16 + 8 * lg;
      const int gn = block_n + wn * 32 + nt * 16 + lr;
      const float bv = bias[gn];
#pragma unroll
      for (int r = 0; r < 8; ++r)
        C[(size_t)(gm + r) * N + gn] = acc[mt][nt][r] + bv;
    }
  }
}

// ---------------------------------------------------------------------------
// RoPE + repack q/k/v from fp32 qkv [tok][3*1024] into bf16 [B][H][S][64].
// One thread per (token, head, d<32) rotation pair.
// ---------------------------------------------------------------------------
__global__ void rope_pack_kernel(const float* __restrict__ qkv,
                                 unsigned short* __restrict__ qb,
                                 unsigned short* __restrict__ kb,
                                 unsigned short* __restrict__ vb) {
  int idx = blockIdx.x * blockDim.x + threadIdx.x;  // < 4096*16*32
  int d   = idx & 31;
  int h   = (idx >> 5) & 15;
  int tok = idx >> 9;
  int s   = tok & (SEQ - 1);
  int b   = tok >> 11;

  const float* base = qkv + (size_t)tok * N_QKV + h * HEAD_DIM;
  float q1 = base[d],        q2 = base[d + 32];
  float k1 = base[1024 + d], k2 = base[1024 + d + 32];
  float v1 = base[2048 + d], v2 = base[2048 + d + 32];

  // inv_freq = 10000^(-d/32)
  float inv_freq = __expf(-(float)d * (9.210340371976184f / 32.0f));
  float ang = (float)s * inv_freq;
  float sn, cs;
  __sincosf(ang, &sn, &cs);

  size_t o = (((size_t)(b * NUM_HEADS + h) * SEQ) + s) * HEAD_DIM + d;
  qb[o]      = f2bf(q1 * cs - q2 * sn);
  qb[o + 32] = f2bf(q2 * cs + q1 * sn);
  kb[o]      = f2bf(k1 * cs - k2 * sn);
  kb[o + 32] = f2bf(k2 * cs + k1 * sn);
  vb[o]      = f2bf(v1);
  vb[o + 32] = f2bf(v2);
}

// ---------------------------------------------------------------------------
// Flash attention: block = 128 threads = 4 waves, each wave owns one 16-row
// query tile; grid = (S/64, B*H). Streams keys in chunks of 32 with online
// softmax; QK^T and PV both via bf16 WMMA.
// ---------------------------------------------------------------------------
__global__ __launch_bounds__(128) void attn_fwd_kernel(
    const unsigned short* __restrict__ Q, const unsigned short* __restrict__ Km,
    const unsigned short* __restrict__ V, unsigned short* __restrict__ O) {
  __shared__ unsigned short Ks[32][72];      // [key][dim]
  __shared__ unsigned short Vs[64][40];      // transposed: [dim][key]
  __shared__ unsigned short Ps[4][16][40];   // per-wave P staging [qrow][key]

  const int tid  = threadIdx.x;
  const int lane = tid & 31;
  const int wave = tid >> 5;
  const int lg   = lane >> 4;
  const int lr   = lane & 15;

  const int bh = blockIdx.y;            // 0..31
  const int b  = bh >> 4;
  const int h  = bh & 15;
  const int q0 = blockIdx.x * 64 + wave * 16;

  const size_t headbase = (size_t)bh * SEQ * HEAD_DIM;

  // Q fragments for K-chunks d=[0,32) and d=[32,64)
  v16bf aq[2];
  {
    const unsigned short* qrow = Q + headbase + (size_t)(q0 + lr) * HEAD_DIM;
    aq[0] = frag8x2(qrow + 8 * lg);
    aq[1] = frag8x2(qrow + 32 + 8 * lg);
  }

  float m[8], l[8];
  v8f acc[4];
#pragma unroll
  for (int r = 0; r < 8; ++r) { m[r] = -INFINITY; l[r] = 0.0f; }
#pragma unroll
  for (int nt = 0; nt < 4; ++nt) acc[nt] = (v8f)0.0f;

  const int skey = tid >> 2;       // staging: 0..31
  const int sd0  = (tid & 3) * 16;

  for (int kc = 0; kc < SEQ; kc += 32) {
    // Stage K (natural) and V (transposed)
    {
      const unsigned short* ksrc = Km + headbase + (size_t)(kc + skey) * HEAD_DIM + sd0;
      u32x4 a0 = *reinterpret_cast<const u32x4*>(ksrc);
      u32x4 a1 = *reinterpret_cast<const u32x4*>(ksrc + 8);
      *reinterpret_cast<u32x4*>(&Ks[skey][sd0])     = a0;
      *reinterpret_cast<u32x4*>(&Ks[skey][sd0 + 8]) = a1;

      const unsigned short* vsrc = V + headbase + (size_t)(kc + skey) * HEAD_DIM + sd0;
      union { u32x4 q; unsigned short us[8]; } v0, v1;
      v0.q = *reinterpret_cast<const u32x4*>(vsrc);
      v1.q = *reinterpret_cast<const u32x4*>(vsrc + 8);
#pragma unroll
      for (int j = 0; j < 8; ++j) Vs[sd0 + j][skey]     = v0.us[j];
#pragma unroll
      for (int j = 0; j < 8; ++j) Vs[sd0 + 8 + j][skey] = v1.us[j];
    }
    __syncthreads();

    // Scores: S[16 x 32] = Q[16x64] * K^T[64x32], two 16-key halves
    v8f s[2];
    s[0] = (v8f)0.0f;
    s[1] = (v8f)0.0f;
#pragma unroll
    for (int hf = 0; hf < 2; ++hf) {
      const unsigned short* krow = &Ks[hf * 16 + lr][0];
      v16bf b0 = frag8x2(krow + 8 * lg);
      v16bf b1 = frag8x2(krow + 32 + 8 * lg);
      s[hf] = wmma_bf16(aq[0], b0, s[hf]);
      s[hf] = wmma_bf16(aq[1], b1, s[hf]);
    }

    // Online softmax per row (row = r + 8*lg, spread over 16 lanes)
#pragma unroll
    for (int r = 0; r < 8; ++r) {
      float v0 = s[0][r] * 0.125f;   // 1/sqrt(64)
      float v1 = s[1][r] * 0.125f;
      float cm = fmaxf(v0, v1);
#pragma unroll
      for (int mk = 1; mk < 16; mk <<= 1) cm = fmaxf(cm, __shfl_xor(cm, mk, 32));
      float mn = fmaxf(m[r], cm);
      float alpha = __expf(m[r] - mn);
      float e0 = __expf(v0 - mn);
      float e1 = __expf(v1 - mn);
      float rs = e0 + e1;
#pragma unroll
      for (int mk = 1; mk < 16; mk <<= 1) rs += __shfl_xor(rs, mk, 32);
      l[r] = l[r] * alpha + rs;
      m[r] = mn;
#pragma unroll
      for (int nt = 0; nt < 4; ++nt) acc[nt][r] *= alpha;
      int prow = r + 8 * lg;
      Ps[wave][prow][lr]      = f2bf(e0);
      Ps[wave][prow][lr + 16] = f2bf(e1);
    }
    // Per-wave LDS round trip: C-layout -> A-layout transpose of P
    asm volatile("s_wait_dscnt 0" ::: "memory");

    v16bf pf = frag8x2(&Ps[wave][lr][8 * lg]);
#pragma unroll
    for (int nt = 0; nt < 4; ++nt) {
      v16bf bv = frag8x2(&Vs[nt * 16 + lr][8 * lg]);
      acc[nt] = wmma_bf16(pf, bv, acc[nt]);
    }
    __syncthreads();
  }

  // Normalize and write bf16 attention output [B,S,1024]
#pragma unroll
  for (int r = 0; r < 8; ++r) {
    float inv = 1.0f / l[r];
    int srow = q0 + r + 8 * lg;
    size_t obase = ((size_t)(b * SEQ + srow)) * D_MODEL + h * HEAD_DIM;
#pragma unroll
    for (int nt = 0; nt < 4; ++nt)
      O[obase + nt * 16 + lr] = f2bf(acc[nt][r] * inv);
  }
}

// ---------------------------------------------------------------------------
// Host-side orchestration
// ---------------------------------------------------------------------------
extern "C" void kernel_launch(void* const* d_in, const int* in_sizes, int n_in,
                              void* d_out, int out_size, void* d_ws, size_t ws_size,
                              hipStream_t stream) {
  const float* x     = (const float*)d_in[0];
  const float* W_qkv = (const float*)d_in[1];
  const float* b_qkv = (const float*)d_in[2];
  const float* W_out = (const float*)d_in[3];
  const float* b_out = (const float*)d_in[4];
  float* out = (float*)d_out;

  char* ws = (char*)d_ws;
  // Workspace layout (bytes)
  const size_t SZ_XB    = (size_t)M_TOTAL * D_MODEL * 2;        //  8 MB
  const size_t SZ_WQKV  = (size_t)D_MODEL * N_QKV * 2;          //  6 MB
  const size_t SZ_WOUT  = (size_t)D_MODEL * D_MODEL * 2;        //  2 MB
  const size_t SZ_QKVF  = (size_t)M_TOTAL * N_QKV * 4;          // 48 MB
  const size_t SZ_HEAD  = (size_t)BATCH * NUM_HEADS * SEQ * HEAD_DIM * 2; // 8 MB

  size_t off = 0;
  unsigned short* xb    = (unsigned short*)(ws + off); off += SZ_XB;
  unsigned short* wqkvb = (unsigned short*)(ws + off); off += SZ_WQKV;
  unsigned short* woutb = (unsigned short*)(ws + off); off += SZ_WOUT;
  float*          qkvf  = (float*)(ws + off);          off += SZ_QKVF;
  unsigned short* qb    = (unsigned short*)(ws + off); off += SZ_HEAD;
  unsigned short* kb    = (unsigned short*)(ws + off); off += SZ_HEAD;
  unsigned short* vb    = (unsigned short*)(ws + off); off += SZ_HEAD;
  unsigned short* aob   = (unsigned short*)(ws + off); off += SZ_XB;

  // 1) bf16 downconversion
  {
    int n = M_TOTAL * D_MODEL;
    f32_to_bf16_kernel<<<(n + 255) / 256, 256, 0, stream>>>(x, xb, n);
    n = D_MODEL * N_QKV;
    f32_to_bf16_kernel<<<(n + 255) / 256, 256, 0, stream>>>(W_qkv, wqkvb, n);
    n = D_MODEL * D_MODEL;
    f32_to_bf16_kernel<<<(n + 255) / 256, 256, 0, stream>>>(W_out, woutb, n);
  }

  // 2) QKV projection: [4096,1024] x [1024,3072] -> fp32
  gemm_bf16_wmma<<<dim3(N_QKV / 64, M_TOTAL / 128), 256, 0, stream>>>(
      xb, wqkvb, b_qkv, qkvf, M_TOTAL, N_QKV, D_MODEL);

  // 3) RoPE + repack to per-head bf16
  {
    int n = M_TOTAL * NUM_HEADS * 32;
    rope_pack_kernel<<<(n + 255) / 256, 256, 0, stream>>>(qkvf, qb, kb, vb);
  }

  // 4) Flash attention
  attn_fwd_kernel<<<dim3(SEQ / 64, BATCH * NUM_HEADS), 128, 0, stream>>>(
      qb, kb, vb, aob);

  // 5) Output projection: [4096,1024] x [1024,1024] -> fp32 out
  gemm_bf16_wmma<<<dim3(D_MODEL / 64, M_TOTAL / 128), 256, 0, stream>>>(
      aob, woutb, b_out, out, M_TOTAL, D_MODEL, D_MODEL);
}
